// PatternLayer_17265768530806
// MI455X (gfx1250) — compile-verified, matrix operand order checked by reading, not run on hardware
//
#include <hip/hip_runtime.h>
#include <hip/hip_bf16.h>

typedef __attribute__((ext_vector_type(16))) _Float16 v16h;
typedef __attribute__((ext_vector_type(8)))  _Float16 v8h;
typedef __attribute__((ext_vector_type(8)))  float    v8f;

#define DM   1024   // D_MODEL
#define DF   2048   // D_FFN
#define SEQT 2048   // T
#define NB   4      // batch
#define NROW (NB * SEQT)   // 8192
#define WND  64

// ---------------------------------------------------------------------------
// WMMA fragment helpers (CDNA5 16x16x32 f16, wave32).
// Per lane: elems 0..7 = mem[kk + h*8 + 0..7], elems 8..15 = mem[kk+16+h*8+0..7]
// Works for global pointers (global_load_b128) and LDS pointers (ds_load_b128).
// ---------------------------------------------------------------------------
__device__ __forceinline__ v16h load_frag(const _Float16* p) {
  v8h lo = *(const v8h*)(p);
  v8h hi = *(const v8h*)(p + 16);
  v16h r;
#pragma unroll
  for (int i = 0; i < 8; ++i) { r[i] = lo[i]; r[i + 8] = hi[i]; }
  return r;
}

__device__ __forceinline__ v8f wmma_f16(v16h a, v16h b, v8f c) {
  return __builtin_amdgcn_wmma_f32_16x16x32_f16(false, a, false, b, (short)0, c,
                                                false, false);
}

// ---------------------------------------------------------------------------
// Async Global -> LDS staging of an A row-tile (CDNA5 async-memory path).
// The LDS byte address handed to the instruction is derived from the shared
// pointer itself (low 32 bits of the generic shared address == LDS offset).
// The ptrtoint ESCAPES the shared array, so the "memory"-clobbering asm is a
// legal writer of it and the subsequent ds_load reads cannot be folded away.
// Each wave drains its own ASYNCcnt, then the workgroup barrier publishes.
// ---------------------------------------------------------------------------
__device__ __forceinline__ void async_stage_a(_Float16* lds,
                                              const _Float16* __restrict__ g,
                                              int bytes) {
  int t = threadIdx.x;  // blockDim.x == 128
  for (int off = t * 16; off < bytes; off += 128 * 16) {
    unsigned lds_off = (unsigned)(uintptr_t)((char*)lds + off);
    const char* gp = (const char*)g + off;
    asm volatile("global_load_async_to_lds_b128 %0, %1, off"
                 :
                 : "v"(lds_off), "v"(gp)
                 : "memory");
  }
  asm volatile("s_wait_asynccnt 0x0" ::: "memory");
  __syncthreads();
}

// ---------------------------------------------------------------------------
// Weight convert + transpose: Wt[n*K + k] = (f16) W[k*N + n]
// ---------------------------------------------------------------------------
__global__ void cvt_tr_kernel(const float* __restrict__ W,
                              _Float16* __restrict__ Wt, int K, int N) {
  size_t idx = (size_t)blockIdx.x * 256 + threadIdx.x;
  if (idx >= (size_t)K * N) return;
  int n = (int)(idx % N);
  int k = (int)(idx / N);
  Wt[(size_t)n * K + k] = (_Float16)W[idx];
}

// ---------------------------------------------------------------------------
// RMSNorm: one row (D=1024) per block of 256 threads, f32 in -> f16 out
// ---------------------------------------------------------------------------
__global__ void rmsnorm_kernel(const float* __restrict__ x,
                               const float* __restrict__ w,
                               _Float16* __restrict__ xn) {
  __shared__ float red[256];
  int row = blockIdx.x;
  int tid = threadIdx.x;
  const float* xr = x + (size_t)row * DM;
  float ss = 0.f;
#pragma unroll
  for (int i = 0; i < DM / 256; ++i) {
    float v = xr[tid + i * 256];
    ss += v * v;
  }
  red[tid] = ss;
  __syncthreads();
  for (int s = 128; s > 0; s >>= 1) {
    if (tid < s) red[tid] += red[tid + s];
    __syncthreads();
  }
  float inv = __frsqrt_rn(red[0] / (float)DM + 1e-6f);
  _Float16* xo = xn + (size_t)row * DM;
#pragma unroll
  for (int i = 0; i < DM / 256; ++i) {
    int c = tid + i * 256;
    xo[c] = (_Float16)(xr[c] * inv * w[c]);
  }
}

// ---------------------------------------------------------------------------
// QKV: q/k = xn @ Wq/Wk (row-major f16), v stored transposed Vt[b][d][t]
// A-tile staged async into LDS, shared by 4 waves. grid (N/64, M/16), block 128
// ---------------------------------------------------------------------------
__global__ void qkv_kernel(const _Float16* __restrict__ xn,
                           const _Float16* __restrict__ Wtq,
                           const _Float16* __restrict__ Wtk,
                           const _Float16* __restrict__ Wtv,
                           _Float16* __restrict__ q, _Float16* __restrict__ k,
                           _Float16* __restrict__ vt) {
  __shared__ _Float16 As[16 * DM];  // 32 KB
  int wave = threadIdx.x >> 5, lane = threadIdx.x & 31;
  int h = lane >> 4, l15 = lane & 15;
  int mtile = blockIdx.y;
  int ntile = blockIdx.x * 4 + wave;
  async_stage_a(As, xn + (size_t)mtile * 16 * DM, 16 * DM * 2);
  const _Float16* ap = As + l15 * DM + h * 8;  // LDS reads (ds_load_b128)
  const _Float16* bq = Wtq + (size_t)(ntile * 16 + l15) * DM + h * 8;
  const _Float16* bk = Wtk + (size_t)(ntile * 16 + l15) * DM + h * 8;
  const _Float16* bv = Wtv + (size_t)(ntile * 16 + l15) * DM + h * 8;
  v8f cq = {}, ck = {}, cv = {};
  for (int kk = 0; kk < DM; kk += 32) {
    __builtin_prefetch(bq + kk + 128, 0, 3);
    v16h a = load_frag(ap + kk);
    cq = wmma_f16(a, load_frag(bq + kk), cq);
    ck = wmma_f16(a, load_frag(bk + kk), ck);
    cv = wmma_f16(a, load_frag(bv + kk), cv);
  }
  int col = ntile * 16 + l15;
#pragma unroll
  for (int v = 0; v < 8; ++v) {
    int r = mtile * 16 + v + 8 * h;
    q[(size_t)r * DM + col] = (_Float16)cq[v];
    k[(size_t)r * DM + col] = (_Float16)ck[v];
    int batch = r >> 11, tloc = r & (SEQT - 1);
    vt[((size_t)batch * DM + col) * SEQT + tloc] = (_Float16)cv[v];
  }
}

// ---------------------------------------------------------------------------
// Attention pass 1: per 16-query tile, S = Q K^T over 5 key tiles (window 64),
// masked softmax over 80 candidate keys, store P (f16) padded to 96.
// ---------------------------------------------------------------------------
__global__ void attn_p_kernel(const _Float16* __restrict__ q,
                              const _Float16* __restrict__ k,
                              _Float16* __restrict__ P) {
  int wave = threadIdx.x >> 5, lane = threadIdx.x & 31;
  int g = blockIdx.x * 4 + wave;          // 0..511 global q-tile
  int batch = g >> 7, qt = g & 127;       // 128 q-tiles per batch
  int h = lane >> 4, l15 = lane & 15;
  const _Float16* ap = q + ((size_t)g * 16 + l15) * DM + h * 8;
  float s[5][8];
#pragma unroll
  for (int t = 0; t < 5; ++t) {
    int kt = qt - 4 + t;
    if (kt < 0) {
#pragma unroll
      for (int v = 0; v < 8; ++v) s[t][v] = -1e30f;
      continue;
    }
    const _Float16* bp =
        k + ((size_t)batch * SEQT + kt * 16 + l15) * DM + h * 8;
    v8f c = {};
    for (int kk = 0; kk < DM; kk += 32)
      c = wmma_f16(load_frag(ap + kk), load_frag(bp + kk), c);
    const float scale = 0.03125f;  // 1/sqrt(1024)
#pragma unroll
    for (int v = 0; v < 8; ++v) {
      int i = qt * 16 + v + 8 * h;
      int j = kt * 16 + l15;
      int d = i - j;
      s[t][v] = (d >= 0 && d < WND) ? c[v] * scale : -1e30f;
    }
  }
  _Float16* pb = P + (size_t)g * 16 * 96;
#pragma unroll
  for (int v = 0; v < 8; ++v) {
    float m = s[0][v];
#pragma unroll
    for (int t = 1; t < 5; ++t) m = fmaxf(m, s[t][v]);
#pragma unroll
    for (int off = 8; off > 0; off >>= 1)
      m = fmaxf(m, __shfl_xor(m, off, 16));
    float e = 0.f;
#pragma unroll
    for (int t = 0; t < 5; ++t) {
      s[t][v] = __expf(s[t][v] - m);
      e += s[t][v];
    }
#pragma unroll
    for (int off = 8; off > 0; off >>= 1) e += __shfl_xor(e, off, 16);
    float inv = 1.f / e;
    int mrow = v + 8 * h;
#pragma unroll
    for (int t = 0; t < 5; ++t)
      pb[mrow * 96 + t * 16 + l15] = (_Float16)(s[t][v] * inv);
    pb[mrow * 96 + 80 + l15] = (_Float16)0.f;  // pad slots 80..95
  }
}

// ---------------------------------------------------------------------------
// Attention pass 2: attended = P(16x96) @ V(96xD) using Vt[b][d][t].
// grid (D/64, 512), block 128. Key range may start below 0 -> guarded chunks.
// ---------------------------------------------------------------------------
__global__ void attn_pv_kernel(const _Float16* __restrict__ P,
                               const _Float16* __restrict__ vt,
                               _Float16* __restrict__ att) {
  int wave = threadIdx.x >> 5, lane = threadIdx.x & 31;
  int g = blockIdx.y;
  int dt = blockIdx.x * 4 + wave;
  int batch = g >> 7, qt = g & 127;
  int h = lane >> 4, l15 = lane & 15;
  const _Float16* ap = P + (size_t)g * 16 * 96 + l15 * 96 + h * 8;
  const _Float16* vrow = vt + ((size_t)batch * DM + dt * 16 + l15) * SEQT;
  int jbase = qt * 16 - 64;
  v8f c = {};
#pragma unroll
  for (int kk = 0; kk < 96; kk += 32) {
    v16h a = load_frag(ap + kk);
    int c0 = jbase + kk + h * 8;
    int c1 = c0 + 16;
    bool ok0 = (c0 >= 0) && (c0 + 7 < SEQT);
    bool ok1 = (c1 >= 0) && (c1 + 7 < SEQT);
    v16h b;
#pragma unroll
    for (int i = 0; i < 8; ++i) {
      b[i]     = ok0 ? vrow[c0 + i] : (_Float16)0.f;
      b[i + 8] = ok1 ? vrow[c1 + i] : (_Float16)0.f;
    }
    c = wmma_f16(a, b, c);
  }
#pragma unroll
  for (int v = 0; v < 8; ++v) {
    int r = g * 16 + v + 8 * h;
    att[(size_t)r * DM + dt * 16 + l15] = (_Float16)c[v];
  }
}

// ---------------------------------------------------------------------------
// x2 = x + attended @ Wo (f32 out, residual). LDS-staged A, 2 n-tiles/wave.
// grid (DM/128, NROW/16), block 128
// ---------------------------------------------------------------------------
__global__ void wo_kernel(const _Float16* __restrict__ att,
                          const _Float16* __restrict__ Wto,
                          const float* __restrict__ x,
                          float* __restrict__ x2) {
  __shared__ _Float16 As[16 * DM];  // 32 KB
  int wave = threadIdx.x >> 5, lane = threadIdx.x & 31;
  int h = lane >> 4, l15 = lane & 15;
  int mtile = blockIdx.y;
  int ntb = (blockIdx.x * 4 + wave) * 2;
  async_stage_a(As, att + (size_t)mtile * 16 * DM, 16 * DM * 2);
  const _Float16* ap = As + l15 * DM + h * 8;
  const _Float16* b0 = Wto + (size_t)(ntb * 16 + l15) * DM + h * 8;
  const _Float16* b1 = Wto + (size_t)((ntb + 1) * 16 + l15) * DM + h * 8;
  v8f c0 = {}, c1 = {};
  for (int kk = 0; kk < DM; kk += 32) {
    __builtin_prefetch(b0 + kk + 128, 0, 3);
    v16h a = load_frag(ap + kk);
    c0 = wmma_f16(a, load_frag(b0 + kk), c0);
    c1 = wmma_f16(a, load_frag(b1 + kk), c1);
  }
#pragma unroll
  for (int v = 0; v < 8; ++v) {
    size_t r = (size_t)(mtile * 16 + v + 8 * h);
    int col0 = ntb * 16 + l15;
    x2[r * DM + col0] = x[r * DM + col0] + c0[v];
    x2[r * DM + col0 + 16] = x[r * DM + col0 + 16] + c1[v];
  }
}

// ---------------------------------------------------------------------------
// FFN in: hh = silu(xn2 @ W1) * (xn2 @ W3), f16 out [8192 x 2048].
// LDS-staged A. grid (DF/64, NROW/16), block 128
// ---------------------------------------------------------------------------
__global__ void ffn1_kernel(const _Float16* __restrict__ xn2,
                            const _Float16* __restrict__ Wt1,
                            const _Float16* __restrict__ Wt3,
                            _Float16* __restrict__ hh) {
  __shared__ _Float16 As[16 * DM];  // 32 KB
  int wave = threadIdx.x >> 5, lane = threadIdx.x & 31;
  int h = lane >> 4, l15 = lane & 15;
  int mtile = blockIdx.y;
  int ntile = blockIdx.x * 4 + wave;
  async_stage_a(As, xn2 + (size_t)mtile * 16 * DM, 16 * DM * 2);
  const _Float16* ap = As + l15 * DM + h * 8;
  const _Float16* b1 = Wt1 + (size_t)(ntile * 16 + l15) * DM + h * 8;
  const _Float16* b3 = Wt3 + (size_t)(ntile * 16 + l15) * DM + h * 8;
  v8f cg = {}, cu = {};
  for (int kk = 0; kk < DM; kk += 32) {
    __builtin_prefetch(b1 + kk + 128, 0, 3);
    v16h a = load_frag(ap + kk);
    cg = wmma_f16(a, load_frag(b1 + kk), cg);
    cu = wmma_f16(a, load_frag(b3 + kk), cu);
  }
  int col = ntile * 16 + l15;
#pragma unroll
  for (int v = 0; v < 8; ++v) {
    size_t r = (size_t)(mtile * 16 + v + 8 * h);
    float gv = cg[v];
    float sl = gv * (1.f / (1.f + __expf(-gv)));   // silu
    hh[r * DF + col] = (_Float16)(sl * cu[v]);
  }
}

// ---------------------------------------------------------------------------
// FFN out + residual: out = x2 + hh @ W2 (K=2048, f32 out).
// LDS-staged A (64 KB), 2 n-tiles/wave. grid (DM/128, NROW/16), block 128
// ---------------------------------------------------------------------------
__global__ void ffn2_kernel(const _Float16* __restrict__ hh,
                            const _Float16* __restrict__ Wt2,
                            const float* __restrict__ x2,
                            float* __restrict__ out) {
  __shared__ _Float16 As[16 * DF];  // 64 KB
  int wave = threadIdx.x >> 5, lane = threadIdx.x & 31;
  int h = lane >> 4, l15 = lane & 15;
  int mtile = blockIdx.y;
  int ntb = (blockIdx.x * 4 + wave) * 2;
  async_stage_a(As, hh + (size_t)mtile * 16 * DF, 16 * DF * 2);
  const _Float16* ap = As + l15 * DF + h * 8;
  const _Float16* b0 = Wt2 + (size_t)(ntb * 16 + l15) * DF + h * 8;
  const _Float16* b1 = Wt2 + (size_t)((ntb + 1) * 16 + l15) * DF + h * 8;
  v8f c0 = {}, c1 = {};
  for (int kk = 0; kk < DF; kk += 32) {
    __builtin_prefetch(b0 + kk + 128, 0, 3);
    v16h a = load_frag(ap + kk);
    c0 = wmma_f16(a, load_frag(b0 + kk), c0);
    c1 = wmma_f16(a, load_frag(b1 + kk), c1);
  }
#pragma unroll
  for (int v = 0; v < 8; ++v) {
    size_t r = (size_t)(mtile * 16 + v + 8 * h);
    int col0 = ntb * 16 + l15;
    out[r * DM + col0] = x2[r * DM + col0] + c0[v];
    out[r * DM + col0 + 16] = x2[r * DM + col0 + 16] + c1[v];
  }
}

// ---------------------------------------------------------------------------
extern "C" void kernel_launch(void* const* d_in, const int* in_sizes, int n_in,
                              void* d_out, int out_size, void* d_ws,
                              size_t ws_size, hipStream_t stream) {
  const float* x       = (const float*)d_in[0];
  const float* norm1_w = (const float*)d_in[1];
  const float* norm2_w = (const float*)d_in[2];
  const float* Wq = (const float*)d_in[3];
  const float* Wk = (const float*)d_in[4];
  const float* Wv = (const float*)d_in[5];
  const float* Wo = (const float*)d_in[6];
  const float* W1 = (const float*)d_in[7];
  const float* W2 = (const float*)d_in[8];
  const float* W3 = (const float*)d_in[9];
  float* out = (float*)d_out;

  char* ws = (char*)d_ws;
  const size_t MB = 1ull << 20;
  _Float16* Wtq = (_Float16*)(ws + 0 * MB);    // 2 MB  (1024x1024)
  _Float16* Wtk = (_Float16*)(ws + 2 * MB);    // 2 MB
  _Float16* Wtv = (_Float16*)(ws + 4 * MB);    // 2 MB
  _Float16* Wto = (_Float16*)(ws + 6 * MB);    // 2 MB
  _Float16* Wt1 = (_Float16*)(ws + 8 * MB);    // 4 MB  (N=2048,K=1024)
  _Float16* Wt3 = (_Float16*)(ws + 12 * MB);   // 4 MB
  _Float16* Wt2 = (_Float16*)(ws + 16 * MB);   // 4 MB  (N=1024,K=2048)
  _Float16* xn  = (_Float16*)(ws + 20 * MB);   // 16 MB (also reused as xn2)
  _Float16* q   = (_Float16*)(ws + 36 * MB);   // 16 MB
  _Float16* kb  = (_Float16*)(ws + 52 * MB);   // 16 MB
  _Float16* vt  = (_Float16*)(ws + 68 * MB);   // 16 MB
  _Float16* P   = (_Float16*)(ws + 84 * MB);   // 1.5 MB (8192x96)
  _Float16* att = (_Float16*)(ws + 86 * MB);   // 16 MB
  float*    x2  = (float*)   (ws + 102 * MB);  // 32 MB
  _Float16* hh  = q;                           // reuse q+k (32 MB) after attn
  _Float16* xn2 = xn;                          // reuse xn after QKV

  // 1) weight convert + transpose to f16
  cvt_tr_kernel<<<(DM * DM + 255) / 256, 256, 0, stream>>>(Wq, Wtq, DM, DM);
  cvt_tr_kernel<<<(DM * DM + 255) / 256, 256, 0, stream>>>(Wk, Wtk, DM, DM);
  cvt_tr_kernel<<<(DM * DM + 255) / 256, 256, 0, stream>>>(Wv, Wtv, DM, DM);
  cvt_tr_kernel<<<(DM * DM + 255) / 256, 256, 0, stream>>>(Wo, Wto, DM, DM);
  cvt_tr_kernel<<<(DM * DF + 255) / 256, 256, 0, stream>>>(W1, Wt1, DM, DF);
  cvt_tr_kernel<<<(DM * DF + 255) / 256, 256, 0, stream>>>(W3, Wt3, DM, DF);
  cvt_tr_kernel<<<(DF * DM + 255) / 256, 256, 0, stream>>>(W2, Wt2, DF, DM);

  // 2) xn = rmsnorm(x) in f16
  rmsnorm_kernel<<<NROW, 256, 0, stream>>>(x, norm1_w, xn);

  // 3) q,k,v projections (v stored transposed)
  qkv_kernel<<<dim3(DM / 64, NROW / 16), 128, 0, stream>>>(xn, Wtq, Wtk, Wtv,
                                                           q, kb, vt);
  // 4) windowed softmax weights P
  attn_p_kernel<<<NROW / 16 / 4, 128, 0, stream>>>(q, kb, P);

  // 5) attended = P @ V
  attn_pv_kernel<<<dim3(DM / 64, NROW / 16), 128, 0, stream>>>(P, vt, att);

  // 6) x2 = x + attended @ Wo
  wo_kernel<<<dim3(DM / 128, NROW / 16), 128, 0, stream>>>(att, Wto, x, x2);

  // 7) xn2 = rmsnorm(x2)
  rmsnorm_kernel<<<NROW, 256, 0, stream>>>(x2, norm2_w, xn2);

  // 8) hh = silu(xn2@W1) * (xn2@W3)
  ffn1_kernel<<<dim3(DF / 64, NROW / 16), 128, 0, stream>>>(xn2, Wt1, Wt3, hh);

  // 9) out = x2 + hh @ W2
  ffn2_kernel<<<dim3(DM / 128, NROW / 16), 128, 0, stream>>>(hh, Wt2, x2, out);
}
